// RNN_LSTM2_1589137900341
// MI455X (gfx1250) — matrix operand order repaired and checked by reading, not compiled, not run on hardware
//
#include <hip/hip_runtime.h>
#include <hip/hip_bf16.h>

// ---------- feature detection (compile-safe guards) ----------
#if defined(__has_builtin)
# if __has_builtin(__builtin_amdgcn_global_load_async_to_lds_b128) && \
     __has_builtin(__builtin_amdgcn_s_wait_asynccnt)
#  define USE_ASYNC 1
# endif
# if __has_builtin(__builtin_amdgcn_tensor_load_to_lds) && \
     __has_builtin(__builtin_amdgcn_s_wait_tensorcnt)
#  define USE_TDM 1
# endif
#endif

// ---------- types ----------
typedef __attribute__((ext_vector_type(16))) __bf16 v16bf;
typedef __attribute__((ext_vector_type(8)))  float  v8f;
typedef __attribute__((ext_vector_type(4)))  unsigned int v4u;
typedef __attribute__((ext_vector_type(4)))  int v4i;
typedef __attribute__((ext_vector_type(8)))  int v8i;

struct U2 { unsigned int x, y; };
struct U4 { unsigned int x, y, z, w; };
union AFrag { U4 u[2]; v16bf v; };

#define AS1 __attribute__((address_space(1)))
#define AS3 __attribute__((address_space(3)))

__device__ __forceinline__ AS3 v4i* to_lds128(const void* p) {
    // generic LDS address: low 32 bits are the LDS byte offset (ISA aperture rule)
    return (AS3 v4i*)(unsigned int)(uintptr_t)p;
}
__device__ __forceinline__ AS1 v4i* to_gbl128(const void* p) {
    return (AS1 v4i*)(uintptr_t)p;
}

__device__ __forceinline__ unsigned short f2bf(float f) {
    unsigned int u = __builtin_bit_cast(unsigned int, f);
    unsigned int r = u + 0x7FFFu + ((u >> 16) & 1u);   // RNE
    return (unsigned short)(r >> 16);
}
__device__ __forceinline__ float sigmoidf_(float x) {
    return 1.0f / (1.0f + __expf(-x));
}

// Problem constants
#define BB 64
#define TT 256
#define DD 512
#define HH 1024
#define KTOT 1536   // D + H
#define NCHUNK (KTOT / 32)

// ---------- prep kernels (one-time, parallel) ----------
__global__ void prep_state(unsigned short* hbuf, float* cbuf) {
    int i = blockIdx.x * 256 + threadIdx.x;   // B*H = 65536
    hbuf[i] = 0;
    cbuf[i] = 0.0f;
}

// wcat[jt][gate][row][1536] bf16 : cols 0..511 from U, 512..1535 from W
__global__ void prep_wcat(const float* __restrict__ uf_w, const float* __restrict__ ug_w,
                          const float* __restrict__ uc_w, const float* __restrict__ uo_w,
                          const float* __restrict__ wf_w, const float* __restrict__ wg_w,
                          const float* __restrict__ wc_w, const float* __restrict__ wo_w,
                          unsigned short* __restrict__ wcat) {
    const int r    = blockIdx.x;                 // 0..4095
    const int gate = (r >> 6) & 3;
    const int jg   = ((r >> 8) << 6) | (r & 63); // global H row
    const float* U = (gate == 0) ? uf_w : (gate == 1) ? ug_w : (gate == 2) ? uc_w : uo_w;
    const float* W = (gate == 0) ? wf_w : (gate == 1) ? wg_w : (gate == 2) ? wc_w : wo_w;
    unsigned short* dst = wcat + (size_t)r * KTOT;
    for (int k = threadIdx.x; k < KTOT; k += 256) {
        float v = (k < DD) ? U[(size_t)jg * DD + k] : W[(size_t)jg * HH + (k - DD)];
        dst[k] = f2bf(v);
    }
}

// xbf[t][m][D] bf16 from x[m][t][D] f32
__global__ void prep_x(const float* __restrict__ x, unsigned short* __restrict__ xbf) {
    const int b = blockIdx.x;        // t*64 + m, 16384 blocks
    const int t = b >> 6, m = b & 63;
    const float* src = x + ((size_t)m * TT + t) * DD;
    unsigned short* dst = xbf + (size_t)b * DD;
    for (int k = threadIdx.x; k < DD; k += 256) dst[k] = f2bf(src[k]);
}

__global__ void prep_ow(const float* __restrict__ o_w, unsigned short* __restrict__ owbf) {
    const size_t i = ((size_t)blockIdx.x * 256 + threadIdx.x) * 4;  // 512 blocks
    float4 f = *(const float4*)(o_w + i);
    owbf[i + 0] = f2bf(f.x); owbf[i + 1] = f2bf(f.y);
    owbf[i + 2] = f2bf(f.z); owbf[i + 3] = f2bf(f.w);
}

__global__ void prep_bias(const float* __restrict__ uf_b, const float* __restrict__ ug_b,
                          const float* __restrict__ uc_b, const float* __restrict__ uo_b,
                          const float* __restrict__ wf_b, const float* __restrict__ wg_b,
                          const float* __restrict__ wc_b, const float* __restrict__ wo_b,
                          float* __restrict__ bcat) {
    const int i = blockIdx.x * 256 + threadIdx.x;  // 0..4095
    const int g = i >> 10, j = i & 1023;
    const float* U = (g == 0) ? uf_b : (g == 1) ? ug_b : (g == 2) ? uc_b : uo_b;
    const float* W = (g == 0) ? wf_b : (g == 1) ? wg_b : (g == 2) ? wc_b : wo_b;
    bcat[i] = U[j] + W[j];
}

// ---------- per-timestep fused gate-GEMM + recurrence ----------
struct StepSmem {
    union {
        struct {
            unsigned short A[2][64][32];       // 8 KB  : double-buffered A tile
            unsigned short Bt[2][4][64][32];   // 32 KB : double-buffered weight tile
        } s;
        float preact[4][64][64];               // 64 KB : gate pre-activations (f32)
    };
};

__global__ __launch_bounds__(256)
void lstm_step(int t, const unsigned short* __restrict__ xbf,
               const unsigned short* __restrict__ wcat,
               const float* __restrict__ bcat,
               unsigned short* __restrict__ hbuf, float* __restrict__ cbuf) {
    __shared__ __attribute__((aligned(32))) StepSmem sm;

    const int tid  = threadIdx.x;
    const int lane = tid & 31;
    const int wave = tid >> 5;          // 0..7
    const int j0   = blockIdx.x * 64;   // H-slice base (16 blocks)

    const int gate = wave >> 1;         // 0..3 : f,g,c,o
    const int nh   = wave & 1;          // n-half of 64-wide slice

    // this block's fused weight region: contiguous [256][1536] bf16
    const unsigned short* wblk = wcat + (size_t)blockIdx.x * 256 * KTOT;

    // A loader: row = tid>>2 (batch), 8 consecutive K per thread
    const int am = tid >> 2;
    const int ak = (tid & 3) * 8;
    const unsigned short* xrow = xbf + ((size_t)t * BB + am) * DD;
    const unsigned short* hrow = hbuf + ((size_t)t * BB + am) * HH;

#if defined(USE_TDM)
    const unsigned int ldsBt0 = (unsigned int)(uintptr_t)&sm.s.Bt[0][0][0][0];
#endif

    // Issue the (TDM + async) transfers for K-chunk kcc into buffer kcc&1.
    auto issue_loads = [&](int kcc) {
        const int kg  = kcc * 32;
        const int buf = kcc & 1;
        const unsigned short* asrc =
            (kg < DD) ? (xrow + kg + ak) : (hrow + (kg - DD) + ak);
#if defined(USE_ASYNC)
        __builtin_amdgcn_global_load_async_to_lds_b128(
            to_gbl128(asrc), to_lds128(&sm.s.A[buf][am][ak]), 0, 0);
#else
        *(U4*)&sm.s.A[buf][am][ak] = *(const U4*)asrc;
#endif
#if defined(USE_TDM)
        if (wave == 0) {
            const unsigned long long ga =
                (unsigned long long)(uintptr_t)wblk + (unsigned long long)kg * 2u;
            v4u g0 = { 1u,                             // count=1, no gather
                       ldsBt0 + (unsigned)buf * 16384u,// lds_addr (bytes)
                       (unsigned int)ga,               // global_addr[31:0]
                       (unsigned int)((ga >> 32) & 0x1FFFFFFu) | (2u << 30) }; // type=2
            v8i g1 = { (int)(1u << 16),       // data_size = 2 bytes
                       (int)(1536u << 16),    // tensor_dim0[15:0] << 16
                       (int)(256u << 16),     // dim0 hi=0 | tensor_dim1[15:0] << 16
                       (int)(32u << 16),      // dim1 hi=0 | tile_dim0=32
                       (int)(256u),           // tile_dim1=256, tile_dim2=0
                       (int)(1536u),          // tensor_dim0_stride lo
                       0, 0 };
            v4i gz = {};
#if __clang_major__ >= 23
            v8i gz8 = {};
            __builtin_amdgcn_tensor_load_to_lds(g0, g1, gz, gz, gz8, 0);
#else
            __builtin_amdgcn_tensor_load_to_lds(g0, g1, gz, gz, 0);
#endif
        }
#else
        {   // fallback: one row per thread, pure bf16 copy
            const unsigned short* src = wblk + (size_t)tid * KTOT + kg;
            unsigned short* dst = &sm.s.Bt[buf][0][0][0] + (size_t)tid * 32;
#pragma unroll
            for (int i = 0; i < 32; i += 8)
                *(U4*)(dst + i) = *(const U4*)(src + i);
        }
#endif
    };

    v8f acc[4][2] = {};

    issue_loads(0);   // prologue

    for (int kc = 0; kc < NCHUNK; ++kc) {
        // Prefetch next chunk into the other buffer, then wait for the
        // *previous* transfer only (counters complete in order per wave).
        if (kc + 1 < NCHUNK) {
            issue_loads(kc + 1);
#if defined(USE_ASYNC)
            __builtin_amdgcn_s_wait_asynccnt(1);
#endif
#if defined(USE_TDM)
            __builtin_amdgcn_s_wait_tensorcnt(1);
#endif
        } else {
#if defined(USE_ASYNC)
            __builtin_amdgcn_s_wait_asynccnt(0);
#endif
#if defined(USE_TDM)
            __builtin_amdgcn_s_wait_tensorcnt(0);
#endif
        }
        __syncthreads();

        // ---- WMMA: 4 M-frags x 2 N-frags per wave ----
        const int buf = kc & 1;
        AFrag fa[4];
        const int kb = (lane >> 4) * 8;
#pragma unroll
        for (int mf = 0; mf < 4; ++mf) {
            const int m = mf * 16 + (lane & 15);
            fa[mf].u[0] = *(const U4*)&sm.s.A[buf][m][kb];
            fa[mf].u[1] = *(const U4*)&sm.s.A[buf][m][kb + 16];
        }
        const int khalf = (lane >> 4) * 8;
        v16bf fb[2];
#pragma unroll
        for (int nf = 0; nf < 2; ++nf) {
            const int n = nh * 32 + nf * 16 + (lane & 15);
            AFrag tb;
            tb.u[0] = *(const U4*)&sm.s.Bt[buf][gate][n][khalf];
            tb.u[1] = *(const U4*)&sm.s.Bt[buf][gate][n][khalf + 16];
            fb[nf] = tb.v;
        }
#pragma unroll
        for (int mf = 0; mf < 4; ++mf)
#pragma unroll
            for (int nf = 0; nf < 2; ++nf)
                acc[mf][nf] = __builtin_amdgcn_wmma_f32_16x16x32_bf16(
                    false, fa[mf].v, false, fb[nf], (short)0, acc[mf][nf],
                    false, false);
        __syncthreads();   // readers done before next issue reuses this buffer
    }

    // ---- exchange pre-activations through LDS ----
#pragma unroll
    for (int mf = 0; mf < 4; ++mf) {
#pragma unroll
        for (int nf = 0; nf < 2; ++nf) {
            const int n = nh * 32 + nf * 16 + (lane & 15);
#pragma unroll
            for (int v = 0; v < 8; ++v) {
                const int m = mf * 16 + v + 8 * (lane >> 4);
                sm.preact[gate][m][n] = acc[mf][nf][v];
            }
        }
    }
    __syncthreads();

    // ---- element-wise recurrence: 16 elements per thread ----
    const int em = tid >> 2;
    const int jb = (tid & 3) * 16;
#pragma unroll 4
    for (int i = 0; i < 16; ++i) {
        const int j  = jb + i;
        const int jg = j0 + j;
        const float pf = sm.preact[0][em][j] + bcat[jg];
        const float pg = sm.preact[1][em][j] + bcat[HH + jg];
        const float pc = sm.preact[2][em][j] + bcat[2 * HH + jg];
        const float po = sm.preact[3][em][j] + bcat[3 * HH + jg];
        const float f  = sigmoidf_(pf);
        const float g  = sigmoidf_(pg);
        const float cd = tanhf(pc);
        const float co = cbuf[(size_t)em * HH + jg];
        const float cn = f * co + g * cd;
        const float h  = tanhf(cn) * sigmoidf_(po);
        cbuf[(size_t)em * HH + jg] = cn;
        hbuf[((size_t)(t + 1) * BB + em) * HH + jg] = f2bf(h);
    }
}

// ---------- output projection: sigmoid(hs @ o_w^T + o_b) ----------
__global__ __launch_bounds__(256)
void out_proj(const unsigned short* __restrict__ hbuf,
              const unsigned short* __restrict__ owbf, const float* __restrict__ o_b,
              float* __restrict__ out) {
    __shared__ __attribute__((aligned(32))) unsigned short sA[32][32];   // 2 KB
    __shared__ __attribute__((aligned(32))) unsigned short sB[256][32];  // 16 KB

    const int tid   = threadIdx.x;
    const int lane  = tid & 31;
    const int wave  = tid >> 5;
    const int waveM = wave >> 2;   // 0..1
    const int waveN = wave & 3;    // 0..3
    const int mbase = blockIdx.x * 32;    // 512 blocks over M = B*T
    const int nbase = blockIdx.y * 256;   // 2 blocks over N = D

    const int arow = tid >> 3, ak = (tid & 7) * 4;       // A: 32x32
    const unsigned short* pwrow = owbf + (size_t)(nbase + tid) * HH;

    v8f acc[4] = {};

    for (int kc = 0; kc < HH / 32; ++kc) {
        const int kglob = kc * 32;
        // A: h-history contiguous bf16: row r maps to hbuf[(r+64)*H]
        *(U2*)&sA[arow][ak] =
            *(const U2*)(hbuf + ((size_t)(mbase + arow) + BB) * HH + kglob + ak);
        // B: one bf16 o_w row per thread
#pragma unroll
        for (int i = 0; i < 32; i += 8)
            *(U4*)&sB[tid][i] = *(const U4*)(pwrow + kglob + i);
        if (kc + 1 < HH / 32) __builtin_prefetch(pwrow + kglob + 32, 0, 0);
        __syncthreads();

        AFrag fa;
        const int kb = (lane >> 4) * 8;
        const int mrow = waveM * 16 + (lane & 15);
        fa.u[0] = *(const U4*)&sA[mrow][kb];
        fa.u[1] = *(const U4*)&sA[mrow][kb + 16];
        const int khalf = (lane >> 4) * 8;
#pragma unroll
        for (int nf = 0; nf < 4; ++nf) {
            const int n = waveN * 64 + nf * 16 + (lane & 15);
            AFrag tb;
            tb.u[0] = *(const U4*)&sB[n][khalf];
            tb.u[1] = *(const U4*)&sB[n][khalf + 16];
            v16bf fb = tb.v;
            acc[nf] = __builtin_amdgcn_wmma_f32_16x16x32_bf16(
                false, fa.v, false, fb, (short)0, acc[nf], false, false);
        }
        __syncthreads();
    }

#pragma unroll
    for (int nf = 0; nf < 4; ++nf) {
        const int n = nbase + waveN * 64 + nf * 16 + (lane & 15);
        const float bias = o_b[n];
#pragma unroll
        for (int v = 0; v < 8; ++v) {
            const int r  = mbase + waveM * 16 + v + 8 * (lane >> 4); // t*64+mb
            const int tt = r >> 6;
            const int mb = r & 63;
            out[((size_t)mb * TT + tt) * DD + n] = sigmoidf_(acc[nf][v] + bias);
        }
    }
}

// ---------- launch ----------
extern "C" void kernel_launch(void* const* d_in, const int* in_sizes, int n_in,
                              void* d_out, int out_size, void* d_ws, size_t ws_size,
                              hipStream_t stream) {
    const float* x    = (const float*)d_in[0];
    const float* uf_w = (const float*)d_in[1];
    const float* uf_b = (const float*)d_in[2];
    const float* ug_w = (const float*)d_in[3];
    const float* ug_b = (const float*)d_in[4];
    const float* u_w  = (const float*)d_in[5];
    const float* u_b  = (const float*)d_in[6];
    const float* uo_w = (const float*)d_in[7];
    const float* uo_b = (const float*)d_in[8];
    const float* wf_w = (const float*)d_in[9];
    const float* wf_b = (const float*)d_in[10];
    const float* wg_w = (const float*)d_in[11];
    const float* wg_b = (const float*)d_in[12];
    const float* w_w  = (const float*)d_in[13];
    const float* w_b  = (const float*)d_in[14];
    const float* wo_w = (const float*)d_in[15];
    const float* wo_b = (const float*)d_in[16];
    const float* o_w  = (const float*)d_in[17];
    const float* o_b  = (const float*)d_in[18];

    // Workspace layout (all 256B-aligned):
    char* ws = (char*)d_ws;
    unsigned short* hbuf = (unsigned short*)ws;                       // 33,685,504 B
    float* cbuf          = (float*)(ws + 33685504);                   //    262,144 B
    unsigned short* wcat = (unsigned short*)(ws + 33947648);          // 12,582,912 B
    unsigned short* xbf  = (unsigned short*)(ws + 46530560);          // 16,777,216 B
    unsigned short* owbf = (unsigned short*)(ws + 63307776);          //  1,048,576 B
    float* bcat          = (float*)(ws + 64356352);                   //     16,384 B

    prep_state<<<256, 256, 0, stream>>>(hbuf, cbuf);
    prep_wcat<<<4096, 256, 0, stream>>>(uf_w, ug_w, u_w, uo_w,
                                        wf_w, wg_w, w_w, wo_w, wcat);
    prep_x<<<16384, 256, 0, stream>>>(x, xbf);
    prep_ow<<<512, 256, 0, stream>>>(o_w, owbf);
    prep_bias<<<16, 256, 0, stream>>>(uf_b, ug_b, u_b, uo_b,
                                      wf_b, wg_b, w_b, wo_b, bcat);

    for (int t = 0; t < TT; ++t)
        lstm_step<<<16, 256, 0, stream>>>(t, xbf, wcat, bcat, hbuf, cbuf);

    out_proj<<<dim3(512, 2, 1), 256, 0, stream>>>(hbuf, owbf, o_b, (float*)d_out);
}
